// MambaLayer_19129784336983
// MI455X (gfx1250) — matrix operand order repaired and checked by reading, not compiled, not run on hardware
//
#include <hip/hip_runtime.h>
#include <hip/hip_bf16.h>

#define DIM_C   256
#define L_LEN   4096
#define NB      4
#define D_INNER 512
#define D_STATE 16
#define DT_RANK 16
#define M_ROWS  (NB * L_LEN)      // 16384
#define NCHUNK  32
#define CLEN    (L_LEN / NCHUNK)  // 128

typedef __bf16 bf16_t;
typedef __attribute__((ext_vector_type(16))) __bf16 v16bf;
typedef __attribute__((ext_vector_type(8)))  float  v8f;

union Frag { v16bf v; unsigned u[8]; };

__device__ __forceinline__ bf16_t f2bf(float f) {
  unsigned u = __float_as_uint(f);
  unsigned r = (u + 0x7FFFu + ((u >> 16) & 1u)) >> 16;
  unsigned short s = (unsigned short)r;
  bf16_t o;
  __builtin_memcpy(&o, &s, 2);
  return o;
}

__device__ __forceinline__ float sigm(float x) { return 1.f / (1.f + __expf(-x)); }

// A fragment: 16x32 bf16, row-major source with leading dim lda (elements).
// ISA layout: lane = m + 16*kh ; VGPR0-3 -> K = kh*8 + {0..7}; VGPR4-7 -> K = 16 + kh*8 + {0..7}
__device__ __forceinline__ v16bf load_a_frag(const bf16_t* __restrict__ A, int lda) {
  int lane = threadIdx.x & 31;
  int m = lane & 15, kh = lane >> 4;
  Frag f;
  const bf16_t* base = A + m * lda + kh * 8;
#pragma unroll
  for (int j = 0; j < 4; ++j) f.u[j]     = *(const unsigned*)(base + 2 * j);
#pragma unroll
  for (int j = 0; j < 4; ++j) f.u[4 + j] = *(const unsigned*)(base + 16 + 2 * j);
  return f.v;
}

// B fragment: 32x16 bf16 where B[k][n] = W[n][k], W row-major (N x K), ldw = K.
// ISA layout: lane = n + 16*kh; VGPR0-3 -> K = kh*8 + {0..7}; VGPR4-7 -> K = 16 + kh*8 + {0..7}
__device__ __forceinline__ v16bf load_b_frag(const bf16_t* __restrict__ W, int ldw) {
  int lane = threadIdx.x & 31;
  int n = lane & 15, kb = (lane >> 4) * 8;
  Frag f;
  const bf16_t* base = W + n * ldw + kb;
#pragma unroll
  for (int j = 0; j < 4; ++j) f.u[j]     = *(const unsigned*)(base + 2 * j);
#pragma unroll
  for (int j = 0; j < 4; ++j) f.u[4 + j] = *(const unsigned*)(base + 16 + 2 * j);
  return f.v;
}

// One wave computes a 16(M) x 64(N) tile: 4 accumulators, A fragment reused 4x per K-step.
__device__ __forceinline__ void wave_gemm_16x64(const bf16_t* __restrict__ Arow,
                                                const bf16_t* __restrict__ Wt,
                                                int K, int ksteps, v8f acc[4]) {
  __builtin_prefetch(Wt, 0, 1);  // global_prefetch_b8: pull weight tile toward L2/L0
  for (int ks = 0; ks < ksteps; ++ks) {
    v16bf a = load_a_frag(Arow + ks * 32, K);
#pragma unroll
    for (int i = 0; i < 4; ++i) {
      v16bf bb = load_b_frag(Wt + (i * 16) * K + ks * 32, K);
      acc[i] = __builtin_amdgcn_wmma_f32_16x16x32_bf16(false, a, false, bb,
                                                       (short)0, acc[i], false, false);
    }
  }
}

// ---------------- prep: weights f32 -> bf16, A = -exp(A_log) ----------------
__global__ void k_prep(const float* __restrict__ win, const float* __restrict__ xpw,
                       const float* __restrict__ outw, const float* __restrict__ alog,
                       bf16_t* __restrict__ winb, bf16_t* __restrict__ xpwb,
                       bf16_t* __restrict__ outwb, float* __restrict__ Aneg) {
  int i = blockIdx.x * blockDim.x + threadIdx.x;
  if (i < 1024 * DIM_C)        winb[i]  = f2bf(win[i]);
  if (i < 64 * D_INNER) {
    int r = i >> 9;            // row of padded 64x512
    xpwb[i] = (r < (DT_RANK + 2 * D_STATE)) ? f2bf(xpw[i]) : f2bf(0.f);
  }
  if (i < DIM_C * D_INNER)     outwb[i] = f2bf(outw[i]);
  if (i < D_INNER * D_STATE)   Aneg[i]  = -__expf(alog[i]);
}

// ---------------- LayerNorm over C=256 per (b,l) row ----------------
__global__ void k_layernorm(const float* __restrict__ x, const float* __restrict__ nw,
                            const float* __restrict__ nb, bf16_t* __restrict__ xn) {
  __shared__ float s1[256], s2[256];
  int row = blockIdx.x, c = threadIdx.x;
  int b = row >> 12, l = row & (L_LEN - 1);
  float v = x[((size_t)(b * DIM_C + c)) * L_LEN + l];
  s1[c] = v; s2[c] = v * v;
  __syncthreads();
  for (int st = 128; st > 0; st >>= 1) {
    if (c < st) { s1[c] += s1[c + st]; s2[c] += s2[c + st]; }
    __syncthreads();
  }
  float mu  = s1[0] * (1.f / 256.f);
  float var = s2[0] * (1.f / 256.f) - mu * mu;
  float xh  = (v - mu) * rsqrtf(var + 1e-5f);
  xn[(size_t)row * DIM_C + c] = f2bf(xh * nw[c] + nb[c]);
}

// ---------------- in_proj GEMM: (M x 256) x (1024 x 256)^T ----------------
// epilogue: n<512 -> x_in in (b,d,l) layout (for conv); n>=512 -> silu(z) in (b,l,d)
__global__ void k_gemm_inproj(const bf16_t* __restrict__ Xn, const bf16_t* __restrict__ Win,
                              float* __restrict__ xin, float* __restrict__ zsil) {
  const int K = DIM_C, NT = 1024 / 64, MT = M_ROWS / 16;
  int wave = blockIdx.x * (blockDim.x >> 5) + (threadIdx.x >> 5);
  if (wave >= MT * NT) return;
  int nt = wave % NT, mt = wave / NT;
  v8f z = {0, 0, 0, 0, 0, 0, 0, 0};
  v8f acc[4] = {z, z, z, z};
  wave_gemm_16x64(Xn + (size_t)mt * 16 * K, Win + (size_t)nt * 64 * K, K, K / 32, acc);
  int lane = threadIdx.x & 31, nl = lane & 15, mh = (lane >> 4) * 8;
#pragma unroll
  for (int i = 0; i < 4; ++i) {
#pragma unroll
    for (int r = 0; r < 8; ++r) {
      int row = mt * 16 + mh + r;
      int b = row >> 12, l = row & (L_LEN - 1);
      int n = nt * 64 + i * 16 + nl;
      float v = acc[i][r];
      if (n < D_INNER)
        xin[((size_t)(b * D_INNER + n)) * L_LEN + l] = v;
      else
        zsil[(size_t)row * D_INNER + (n - D_INNER)] = v * sigm(v);
    }
  }
}

// ---------------- depthwise causal conv(4) + SiLU ----------------
__global__ void k_conv_silu(const float* __restrict__ xin, const float* __restrict__ cw,
                            const float* __restrict__ cb, float* __restrict__ uf,
                            bf16_t* __restrict__ ub) {
  int bd = blockIdx.y;                       // b*512 + d
  int d = bd & (D_INNER - 1), b = bd >> 9;
  int l = blockIdx.x * blockDim.x + threadIdx.x;
  const float* src = xin + (size_t)bd * L_LEN;
  float w0 = cw[d * 4 + 0], w1 = cw[d * 4 + 1], w2 = cw[d * 4 + 2], w3 = cw[d * 4 + 3];
  float a = cb[d] + w3 * src[l];
  if (l >= 1) a += w2 * src[l - 1];
  if (l >= 2) a += w1 * src[l - 2];
  if (l >= 3) a += w0 * src[l - 3];
  float u = a * sigm(a);
  size_t o = ((size_t)(b * L_LEN + l)) * D_INNER + d;
  uf[o] = u;
  ub[o] = f2bf(u);
}

// ---------------- x_proj GEMM: (M x 512) x (64 x 512)^T (48 valid, padded) ----------------
__global__ void k_gemm_xproj(const bf16_t* __restrict__ U, const bf16_t* __restrict__ Wxp,
                             float* __restrict__ xdbl) {
  const int K = D_INNER, MT = M_ROWS / 16;
  int wave = blockIdx.x * (blockDim.x >> 5) + (threadIdx.x >> 5);
  if (wave >= MT) return;
  int mt = wave;
  v8f z = {0, 0, 0, 0, 0, 0, 0, 0};
  v8f acc[4] = {z, z, z, z};
  wave_gemm_16x64(U + (size_t)mt * 16 * K, Wxp, K, K / 32, acc);
  int lane = threadIdx.x & 31, nl = lane & 15, mh = (lane >> 4) * 8;
#pragma unroll
  for (int i = 0; i < 4; ++i)
#pragma unroll
    for (int r = 0; r < 8; ++r) {
      int row = mt * 16 + mh + r;
      xdbl[(size_t)row * 64 + i * 16 + nl] = acc[i][r];
    }
}

// ---------------- dt_proj (K=16) + softplus ----------------
__global__ void k_dtproj(const float* __restrict__ xdbl, const float* __restrict__ dtw,
                         const float* __restrict__ dtb, float* __restrict__ dt) {
  __shared__ float r16[DT_RANK];
  int row = blockIdx.x;
  int d = blockIdx.y * blockDim.x + threadIdx.x;
  if (threadIdx.x < DT_RANK) r16[threadIdx.x] = xdbl[(size_t)row * 64 + threadIdx.x];
  __syncthreads();
  float a = dtb[d];
#pragma unroll
  for (int r = 0; r < DT_RANK; ++r) a += r16[r] * dtw[d * DT_RANK + r];
  float sp = (a > 20.f) ? a : log1pf(__expf(a));
  dt[(size_t)row * D_INNER + d] = sp;
}

// ---------------- chunked selective scan ----------------
// pass 1: per (b,d,chunk): local state from h=0 over CLEN steps + sum(dt)
__global__ void k_scan1(const float* __restrict__ dt, const float* __restrict__ uf,
                        const float* __restrict__ xdbl, const float* __restrict__ Aneg,
                        float* __restrict__ hloc, float* __restrict__ dtsum) {
  int idx = blockIdx.x * blockDim.x + threadIdx.x;   // 65536
  int d = idx & (D_INNER - 1);
  int t2 = idx >> 9;
  int c = t2 & (NCHUNK - 1);
  int b = t2 >> 5;
  float A[D_STATE], h[D_STATE];
#pragma unroll
  for (int n = 0; n < D_STATE; ++n) { A[n] = Aneg[d * D_STATE + n]; h[n] = 0.f; }
  float ds = 0.f;
  int rbase = b * L_LEN + c * CLEN;
  for (int t = 0; t < CLEN; ++t) {
    size_t row = (size_t)(rbase + t);
    float dtv = dt[row * D_INNER + d];
    float uv  = uf[row * D_INNER + d];
    float du  = dtv * uv;
    ds += dtv;
    const float* Bp = xdbl + row * 64 + DT_RANK;
#pragma unroll
    for (int n = 0; n < D_STATE; ++n)
      h[n] = h[n] * __expf(dtv * A[n]) + du * Bp[n];
  }
  size_t base = ((size_t)(b * D_INNER + d) * NCHUNK + c) * D_STATE;
#pragma unroll
  for (int n = 0; n < D_STATE; ++n) hloc[base + n] = h[n];
  dtsum[(size_t)(b * D_INNER + d) * NCHUNK + c] = ds;
}

// pass 2: per (b,d,n): 32-step sequential chunk combine; record per-chunk initial state
__global__ void k_scan2(const float* __restrict__ hloc, const float* __restrict__ dtsum,
                        const float* __restrict__ Aneg, float* __restrict__ hinit) {
  int idx = blockIdx.x * blockDim.x + threadIdx.x;   // 32768
  int n = idx & 15;
  int d = (idx >> 4) & (D_INNER - 1);
  int b = idx >> 13;
  float A = Aneg[d * D_STATE + n];
  float h = 0.f;
  size_t bd = (size_t)(b * D_INNER + d);
  for (int c = 0; c < NCHUNK; ++c) {
    size_t o = (bd * NCHUNK + c) * D_STATE + n;
    hinit[o] = h;
    h = h * __expf(A * dtsum[bd * NCHUNK + c]) + hloc[o];
  }
}

// pass 3: replay chunk with correct init state; fuse y = scan + u*D, gate with silu(z), emit bf16
__global__ void k_scan3(const float* __restrict__ dt, const float* __restrict__ uf,
                        const float* __restrict__ xdbl, const float* __restrict__ Aneg,
                        const float* __restrict__ hinit, const float* __restrict__ Dp,
                        const float* __restrict__ zsil, bf16_t* __restrict__ yb) {
  int idx = blockIdx.x * blockDim.x + threadIdx.x;   // 65536
  int d = idx & (D_INNER - 1);
  int t2 = idx >> 9;
  int c = t2 & (NCHUNK - 1);
  int b = t2 >> 5;
  float A[D_STATE], h[D_STATE];
  size_t base = ((size_t)(b * D_INNER + d) * NCHUNK + c) * D_STATE;
#pragma unroll
  for (int n = 0; n < D_STATE; ++n) { A[n] = Aneg[d * D_STATE + n]; h[n] = hinit[base + n]; }
  float Dd = Dp[d];
  int rbase = b * L_LEN + c * CLEN;
  for (int t = 0; t < CLEN; ++t) {
    size_t row = (size_t)(rbase + t);
    float dtv = dt[row * D_INNER + d];
    float uv  = uf[row * D_INNER + d];
    float du  = dtv * uv;
    const float* Bp = xdbl + row * 64 + DT_RANK;
    const float* Cp = xdbl + row * 64 + DT_RANK + D_STATE;
    float y = 0.f;
#pragma unroll
    for (int n = 0; n < D_STATE; ++n) {
      h[n] = h[n] * __expf(dtv * A[n]) + du * Bp[n];
      y += h[n] * Cp[n];
    }
    y = (y + uv * Dd) * zsil[row * D_INNER + d];
    yb[row * D_INNER + d] = f2bf(y);
  }
}

// ---------------- out_proj GEMM: (M x 512) x (256 x 512)^T, transposed store ----------------
__global__ void k_gemm_outproj(const bf16_t* __restrict__ Y, const bf16_t* __restrict__ Wout,
                               float* __restrict__ out) {
  const int K = D_INNER, NT = DIM_C / 64, MT = M_ROWS / 16;
  int wave = blockIdx.x * (blockDim.x >> 5) + (threadIdx.x >> 5);
  if (wave >= MT * NT) return;
  int nt = wave % NT, mt = wave / NT;
  v8f z = {0, 0, 0, 0, 0, 0, 0, 0};
  v8f acc[4] = {z, z, z, z};
  wave_gemm_16x64(Y + (size_t)mt * 16 * K, Wout + (size_t)nt * 64 * K, K, K / 32, acc);
  int lane = threadIdx.x & 31, nl = lane & 15, mh = (lane >> 4) * 8;
#pragma unroll
  for (int i = 0; i < 4; ++i)
#pragma unroll
    for (int r = 0; r < 8; ++r) {
      int row = mt * 16 + mh + r;
      int b = row >> 12, l = row & (L_LEN - 1);
      int n = nt * 64 + i * 16 + nl;
      out[((size_t)(b * DIM_C + n)) * L_LEN + l] = acc[i][r];
    }
}

extern "C" void kernel_launch(void* const* d_in, const int* in_sizes, int n_in,
                              void* d_out, int out_size, void* d_ws, size_t ws_size,
                              hipStream_t stream) {
  const float* x     = (const float*)d_in[0];
  const float* nw    = (const float*)d_in[1];
  const float* nbb   = (const float*)d_in[2];
  const float* winp  = (const float*)d_in[3];
  const float* convw = (const float*)d_in[4];
  const float* convb = (const float*)d_in[5];
  const float* xpw   = (const float*)d_in[6];
  const float* dtw   = (const float*)d_in[7];
  const float* dtb   = (const float*)d_in[8];
  const float* alog  = (const float*)d_in[9];
  const float* Dp    = (const float*)d_in[10];
  const float* outw  = (const float*)d_in[11];
  float* out = (float*)d_out;

  char* p = (char*)d_ws;
  auto take = [&](size_t bytes) -> char* {
    char* r = p;
    p += (bytes + 255) & ~(size_t)255;
    return r;
  };
  bf16_t* winb  = (bf16_t*)take((size_t)1024 * DIM_C * 2);
  bf16_t* xpwb  = (bf16_t*)take((size_t)64 * D_INNER * 2);
  bf16_t* outwb = (bf16_t*)take((size_t)DIM_C * D_INNER * 2);
  float*  Aneg  = (float*) take((size_t)D_INNER * D_STATE * 4);
  bf16_t* xnb   = (bf16_t*)take((size_t)M_ROWS * DIM_C * 2);
  float*  xin   = (float*) take((size_t)NB * D_INNER * L_LEN * 4);
  float*  zsil  = (float*) take((size_t)M_ROWS * D_INNER * 4);
  float*  uf    = (float*) take((size_t)M_ROWS * D_INNER * 4);
  bf16_t* ub    = (bf16_t*)take((size_t)M_ROWS * D_INNER * 2);
  float*  xdbl  = (float*) take((size_t)M_ROWS * 64 * 4);
  float*  dt    = (float*) take((size_t)M_ROWS * D_INNER * 4);
  float*  hloc  = (float*) take((size_t)NB * D_INNER * NCHUNK * D_STATE * 4);
  float*  dtsum = (float*) take((size_t)NB * D_INNER * NCHUNK * 4);
  float*  hinit = (float*) take((size_t)NB * D_INNER * NCHUNK * D_STATE * 4);
  bf16_t* yb    = (bf16_t*)take((size_t)M_ROWS * D_INNER * 2);

  k_prep<<<1024, 256, 0, stream>>>(winp, xpw, outw, alog, winb, xpwb, outwb, Aneg);
  k_layernorm<<<M_ROWS, 256, 0, stream>>>(x, nw, nbb, xnb);
  k_gemm_inproj<<<(M_ROWS / 16) * (1024 / 64) / 8, 256, 0, stream>>>(xnb, winb, xin, zsil);
  k_conv_silu<<<dim3(L_LEN / 256, NB * D_INNER), 256, 0, stream>>>(xin, convw, convb, uf, ub);
  k_gemm_xproj<<<(M_ROWS / 16) / 8, 256, 0, stream>>>(ub, xpwb, xdbl);
  k_dtproj<<<dim3(M_ROWS, D_INNER / 256), 256, 0, stream>>>(xdbl, dtw, dtb, dt);
  k_scan1<<<(NB * D_INNER * NCHUNK) / 256, 256, 0, stream>>>(dt, uf, xdbl, Aneg, hloc, dtsum);
  k_scan2<<<(NB * D_INNER * D_STATE) / 256, 256, 0, stream>>>(hloc, dtsum, Aneg, hinit);
  k_scan3<<<(NB * D_INNER * NCHUNK) / 256, 256, 0, stream>>>(dt, uf, xdbl, Aneg, hinit, Dp, zsil, yb);
  k_gemm_outproj<<<(M_ROWS / 16) * (DIM_C / 64) / 8, 256, 0, stream>>>(yb, outwb, out);
}
